// MultiHeadAttention_23965917511732
// MI455X (gfx1250) — matrix operand order repaired and checked by reading, not compiled, not run on hardware
//
#include <hip/hip_runtime.h>

// ---------------------------------------------------------------------------
// MultiHeadAttention forward for MI455X (gfx1250, wave32, WMMA).
// B=4, N=2048, E=768, H=12, D=64.  All matmuls via v_wmma_f32_16x16x32_f16.
// Flash attention stages K/V tiles in LDS (shared by 8 waves) with
// GLOBAL_LOAD_ASYNC_TO_LDS double-buffering (ASYNCcnt path).
// ---------------------------------------------------------------------------

#define BATCH 4
#define SEQ   2048
#define EMB   768
#define HEADS 12
#define HDIM  64
#define MTOT  (BATCH * SEQ)   // 8192

typedef __attribute__((ext_vector_type(16))) _Float16 v16h;
typedef __attribute__((ext_vector_type(8)))  _Float16 v8h;
typedef __attribute__((ext_vector_type(8)))  float    v8f;
typedef __attribute__((ext_vector_type(4)))  int      v4i;

#if defined(__has_builtin)
#  if __has_builtin(__builtin_amdgcn_global_load_async_to_lds_b128) && \
      __has_builtin(__builtin_amdgcn_s_wait_asynccnt)
#    define HAVE_ASYNC_LDS 1
#  endif
#endif
#ifndef HAVE_ASYNC_LDS
#  define HAVE_ASYNC_LDS 0
#endif

#define CAT16(a, b) __builtin_shufflevector((a), (b), 0,1,2,3,4,5,6,7,8,9,10,11,12,13,14,15)

__device__ __forceinline__ v8f wmma_f16f32(v16h a, v16h b, v8f c) {
  // D = A(16x32) * B(32x16) + C, f32 accumulate
  return __builtin_amdgcn_wmma_f32_16x16x32_f16(false, a, false, b, (short)0, c, false, false);
}

// XOR swizzle within groups of 16 lanes (group-of-32 mode, and=0x1f, or=0)
template <int XORM>
__device__ __forceinline__ float swz_xor(float v) {
  return __int_as_float(
      __builtin_amdgcn_ds_swizzle(__float_as_int(v), 0x001f | (XORM << 10)));
}

// Copy 16 bytes global -> LDS (async if available, else load+ds_store)
__device__ __forceinline__ void g2l_b128(const _Float16* g, _Float16* l) {
#if HAVE_ASYNC_LDS
  __builtin_amdgcn_global_load_async_to_lds_b128(
      (__attribute__((address_space(1))) v4i*)g,
      (__attribute__((address_space(3))) v4i*)l, 0, 0);
#else
  *(v8h*)l = *(const v8h*)g;
#endif
}

__device__ __forceinline__ void wait_stage_and_sync() {
#if HAVE_ASYNC_LDS
  __builtin_amdgcn_s_wait_asynccnt(0);
#endif
  __syncthreads();
}

// ---------------------------------------------------------------------------
// fp32 -> fp16 conversion (n divisible by 4)
// ---------------------------------------------------------------------------
__global__ __launch_bounds__(256) void cvt_f32_to_f16(const float* __restrict__ src,
                                                      _Float16* __restrict__ dst,
                                                      int n) {
  int i = (blockIdx.x * 256 + threadIdx.x) * 4;
  if (i < n) {
    float4 f = *(const float4*)(src + i);
    union { _Float16 h[4]; uint2 u; } p;
    p.h[0] = (_Float16)f.x; p.h[1] = (_Float16)f.y;
    p.h[2] = (_Float16)f.z; p.h[3] = (_Float16)f.w;
    *(uint2*)(dst + i) = p.u;
  }
}

// ---------------------------------------------------------------------------
// WMMA GEMM: Y = X @ W^T + bias.  X:[MTOT,768] f16, W:[768,768] f16 row-major.
// MODE 0: store f16 [MTOT,768]        (Q, K)
// MODE 1: store f16 head-transposed Vt[B][H][D][N]   (V)
// MODE 2: store f32 [MTOT,768]        (final output)
// Block = 256 threads = 8 waves (4 M x 2 N); wave tile 32x32; block 128x64.
// Weights (1.2 MB f16) are L2-resident -> direct global fragment loads.
// ---------------------------------------------------------------------------
template <int MODE>
__global__ __launch_bounds__(256) void wmma_gemm(const _Float16* __restrict__ X,
                                                 const _Float16* __restrict__ W,
                                                 const float* __restrict__ bias,
                                                 void* __restrict__ out) {
  const int K = EMB;
  const int lane = threadIdx.x & 31;
  const int wid  = threadIdx.x >> 5;
  const int lo = lane & 15, hi = lane >> 4;
  const int wm = wid & 3, wn = wid >> 2;
  const int m0 = blockIdx.x * 128 + wm * 32;
  const int n0 = blockIdx.y * 64 + wn * 32;

  v8f acc[2][2] = {};

  for (int k0 = 0; k0 < K; k0 += 32) {
    v16h a[2], bf[2];
#pragma unroll
    for (int i = 0; i < 2; ++i) {
      const _Float16* ap = X + (size_t)(m0 + i * 16 + lo) * K + k0;
      if (k0 + 32 < K) __builtin_prefetch(ap + 32, 0, 1);  // global_prefetch_b8
      v8h a0 = *(const v8h*)(ap + hi * 16);
      v8h a1 = *(const v8h*)(ap + hi * 16 + 8);
      a[i] = CAT16(a0, a1);
    }
#pragma unroll
    for (int j = 0; j < 2; ++j) {
      const _Float16* bp = W + (size_t)(n0 + j * 16 + lo) * K + k0 + hi * 16;
      v8h b0 = *(const v8h*)(bp);
      v8h b1 = *(const v8h*)(bp + 8);
      bf[j] = CAT16(b0, b1);
    }
#pragma unroll
    for (int i = 0; i < 2; ++i)
#pragma unroll
      for (int j = 0; j < 2; ++j)
        acc[i][j] = wmma_f16f32(a[i], bf[j], acc[i][j]);
  }

  // Epilogue.  D layout: VGPR r of lane (lo,hi) = row (sub + r + hi*8), col (n + lo).
#pragma unroll
  for (int j = 0; j < 2; ++j) {
    const int col = n0 + j * 16 + lo;
    const float bv = bias[col];
#pragma unroll
    for (int i = 0; i < 2; ++i) {
      const int mbase = m0 + i * 16 + hi * 8;  // 8 consecutive rows r=0..7
      if constexpr (MODE == 1) {
        // V transposed per head: Vt[b][h][d][tok], 8 consecutive toks -> b128
        union { _Float16 h[8]; uint4 q; } pk;
#pragma unroll
        for (int r = 0; r < 8; ++r) pk.h[r] = (_Float16)(acc[i][j][r] + bv);
        const int hh = col >> 6, dd = col & 63;
        const int bb = mbase >> 11, tok = mbase & (SEQ - 1);
        _Float16* vt = (_Float16*)out;
        *(uint4*)(vt + ((size_t)(bb * HEADS + hh) * HDIM + dd) * SEQ + tok) = pk.q;
      } else {
#pragma unroll
        for (int r = 0; r < 8; ++r) {
          const float v = acc[i][j][r] + bv;
          const size_t off = (size_t)(mbase + r) * EMB + col;
          if constexpr (MODE == 0) ((_Float16*)out)[off] = (_Float16)v;
          else                     ((float*)out)[off] = v;
        }
      }
    }
  }
}

// ---------------------------------------------------------------------------
// Flash attention.  One block = 256 threads = 8 waves, all on one (batch,head);
// each wave owns one 16-row Q tile.  Per 32-key block, K tile (32x64 f16, 4KB)
// and V tile (64x32 f16, 4KB) are staged once into LDS (double-buffered) and
// shared by all 8 waves -> 8x cut in L2/HBM traffic for K/V.
// Q,K in [B,N,E] f16 (head slice contiguous along d), V in Vt[B][H][D][N] f16.
// Online softmax over 32-key blocks; P tile re-laid out D->A via per-wave LDS.
// ---------------------------------------------------------------------------
__global__ __launch_bounds__(256) void flash_attn(const _Float16* __restrict__ Qh,
                                                  const _Float16* __restrict__ Kh,
                                                  const _Float16* __restrict__ Vt,
                                                  _Float16* __restrict__ Ctx) {
  __shared__ _Float16 Ktile[2][32 * 64];   // [key 0..31][d 0..63]
  __shared__ _Float16 Vtile[2][64 * 32];   // [d 0..63][key 0..31]
  __shared__ _Float16 pbuf[8][16 * 32];    // per-wave 16x32 P tile

  const int tid  = threadIdx.x;
  const int lane = tid & 31;
  const int wid  = tid >> 5;
  const int lo = lane & 15, hi = lane >> 4;
  const int bh = blockIdx.x;
  const int b = bh / HEADS, h = bh % HEADS;
  const int q0 = (blockIdx.y * 8 + wid) * 16;

  const _Float16* kbasep = Kh + (size_t)b * SEQ * EMB + h * HDIM;
  const _Float16* vbasep = Vt + (size_t)(b * HEADS + h) * HDIM * SEQ;

  // --- cooperative K/V tile staging (one b128 per thread per tile) ---------
  const int krow = tid >> 3, kch = (tid & 7) * 8;   // 32 rows x 8 chunks
  const int vrow = tid >> 2, vch = (tid & 3) * 8;   // 64 rows x 4 chunks
  auto stage = [&](int kb, int buf) {
    g2l_b128(kbasep + (size_t)(kb + krow) * EMB + kch, &Ktile[buf][krow * 64 + kch]);
    g2l_b128(vbasep + (size_t)vrow * SEQ + kb + vch,   &Vtile[buf][vrow * 32 + vch]);
  };

  // --- Q A-fragments (d 0..31 and 32..63), direct from global --------------
  const _Float16* qp = Qh + (size_t)(b * SEQ + q0 + lo) * EMB + h * HDIM;
  v16h aQ[2];
#pragma unroll
  for (int i = 0; i < 2; ++i) {
    v8h x0 = *(const v8h*)(qp + i * 32 + hi * 16);
    v8h x1 = *(const v8h*)(qp + i * 32 + hi * 16 + 8);
    aQ[i] = CAT16(x0, x1);
  }

  float Mr[8], Lr[8];
  v8f O[4] = {};
#pragma unroll
  for (int r = 0; r < 8; ++r) { Mr[r] = -3.0e38f; Lr[r] = 0.0f; }
  const float c = 0.18033688011112042f;  // (1/sqrt(64)) * log2(e)

  _Float16* pb = pbuf[wid];
  constexpr int NKB = SEQ / 32;  // 64 key blocks

  stage(0, 0);
  wait_stage_and_sync();

  for (int ib = 0; ib < NKB; ++ib) {
    const int buf = ib & 1;
    if (ib + 1 < NKB) stage((ib + 1) * 32, buf ^ 1);  // overlap next transfer

    const _Float16* kt = Ktile[buf];
    const _Float16* vtl = Vtile[buf];

    // K^T B-fragments from LDS: [t] = key subtile (16 keys), [dh] = d half
    v16h bK[2][2];
#pragma unroll
    for (int t = 0; t < 2; ++t) {
      const _Float16* kp = kt + (t * 16 + lo) * 64;
#pragma unroll
      for (int dh = 0; dh < 2; ++dh) {
        v8h x0 = *(const v8h*)(kp + dh * 32 + hi * 16);
        v8h x1 = *(const v8h*)(kp + dh * 32 + hi * 16 + 8);
        bK[t][dh] = CAT16(x0, x1);
      }
    }

    v8f s0 = {}, s1 = {};
    s0 = wmma_f16f32(aQ[0], bK[0][0], s0);
    s0 = wmma_f16f32(aQ[1], bK[0][1], s0);
    s1 = wmma_f16f32(aQ[0], bK[1][0], s1);
    s1 = wmma_f16f32(aQ[1], bK[1][1], s1);

    // Online softmax per row r (row = r + hi*8, spread over 16 lanes)
#pragma unroll
    for (int r = 0; r < 8; ++r) {
      float rv = fmaxf(s0[r], s1[r]);
      rv = fmaxf(rv, swz_xor<1>(rv));
      rv = fmaxf(rv, swz_xor<2>(rv));
      rv = fmaxf(rv, swz_xor<4>(rv));
      rv = fmaxf(rv, swz_xor<8>(rv));
      const float mnew  = fmaxf(Mr[r], rv);
      const float alpha = exp2f((Mr[r] - mnew) * c);
      const float p0 = exp2f((s0[r] - mnew) * c);
      const float p1 = exp2f((s1[r] - mnew) * c);
      float ps = p0 + p1;
      ps += swz_xor<1>(ps);
      ps += swz_xor<2>(ps);
      ps += swz_xor<4>(ps);
      ps += swz_xor<8>(ps);
      Lr[r] = Lr[r] * alpha + ps;
      Mr[r] = mnew;
#pragma unroll
      for (int t = 0; t < 4; ++t) O[t][r] *= alpha;
      // P tile to per-wave LDS in row-major 16x32 (D layout scatter)
      pb[(r + hi * 8) * 32 + lo]      = (_Float16)p0;
      pb[(r + hi * 8) * 32 + 16 + lo] = (_Float16)p1;
    }

    // Re-read P as A fragment (row = lo, K chunks hi*16 and hi*16+8)
    v8h pl0 = *(const v8h*)(pb + lo * 32 + hi * 16);
    v8h pl1 = *(const v8h*)(pb + lo * 32 + hi * 16 + 8);
    v16h aP = CAT16(pl0, pl1);

    // ctx += P(16x32) @ V(32x64): 4 d-subtiles, B frags from LDS V tile
#pragma unroll
    for (int t = 0; t < 4; ++t) {
      const _Float16* vp = vtl + (t * 16 + lo) * 32 + hi * 16;
      v8h x0 = *(const v8h*)(vp);
      v8h x1 = *(const v8h*)(vp + 8);
      v16h bV = CAT16(x0, x1);
      O[t] = wmma_f16f32(aP, bV, O[t]);
    }

    // next buffer's async data landed + everyone done reading current buffer
    wait_stage_and_sync();
  }

  // Normalize and store ctx [B,N,E] f16
  _Float16* cp = Ctx + (size_t)(b * SEQ) * EMB + h * HDIM;
#pragma unroll
  for (int r = 0; r < 8; ++r) {
    const float inv = __builtin_amdgcn_rcpf(Lr[r]);
    const int row = q0 + r + hi * 8;
#pragma unroll
    for (int t = 0; t < 4; ++t)
      cp[(size_t)row * EMB + t * 16 + lo] = (_Float16)(O[t][r] * inv);
  }
}

// ---------------------------------------------------------------------------
// Host launcher
// ---------------------------------------------------------------------------
extern "C" void kernel_launch(void* const* d_in, const int* in_sizes, int n_in,
                              void* d_out, int out_size, void* d_ws, size_t ws_size,
                              hipStream_t stream) {
  (void)in_sizes; (void)n_in; (void)out_size; (void)ws_size;

  const float* x  = (const float*)d_in[0];
  const float* Wq = (const float*)d_in[1];
  const float* bq = (const float*)d_in[2];
  const float* Wk = (const float*)d_in[3];
  const float* bk = (const float*)d_in[4];
  const float* Wv = (const float*)d_in[5];
  const float* bv = (const float*)d_in[6];
  const float* Wo = (const float*)d_in[7];
  const float* bo = (const float*)d_in[8];

  const size_t actBytes = (size_t)MTOT * EMB * sizeof(_Float16);  // 12.6 MB
  const size_t wBytes   = (size_t)EMB * EMB * sizeof(_Float16);   // 1.18 MB

  char* ws = (char*)d_ws;
  size_t off = 0;
  auto carve = [&](size_t bytes) { void* p = ws + off; off += (bytes + 255) & ~(size_t)255; return p; };

  _Float16* xh   = (_Float16*)carve(actBytes);
  _Float16* Wq16 = (_Float16*)carve(wBytes);
  _Float16* Wk16 = (_Float16*)carve(wBytes);
  _Float16* Wv16 = (_Float16*)carve(wBytes);
  _Float16* Wo16 = (_Float16*)carve(wBytes);
  _Float16* Qh   = (_Float16*)carve(actBytes);
  _Float16* Kh   = (_Float16*)carve(actBytes);
  _Float16* Vt   = (_Float16*)carve(actBytes);   // [B][H][D][N]
  _Float16* Ctx  = (_Float16*)carve(actBytes);

  const int nAct = MTOT * EMB;   // 6291456
  const int nW   = EMB * EMB;    // 589824
  cvt_f32_to_f16<<<nAct / 4 / 256, 256, 0, stream>>>(x,  xh,   nAct);
  cvt_f32_to_f16<<<nW   / 4 / 256, 256, 0, stream>>>(Wq, Wq16, nW);
  cvt_f32_to_f16<<<nW   / 4 / 256, 256, 0, stream>>>(Wk, Wk16, nW);
  cvt_f32_to_f16<<<nW   / 4 / 256, 256, 0, stream>>>(Wv, Wv16, nW);
  cvt_f32_to_f16<<<nW   / 4 / 256, 256, 0, stream>>>(Wo, Wo16, nW);

  dim3 gg(MTOT / 128, EMB / 64);  // (64, 12)
  wmma_gemm<0><<<gg, 256, 0, stream>>>(xh, Wq16, bq, (void*)Qh);
  wmma_gemm<0><<<gg, 256, 0, stream>>>(xh, Wk16, bk, (void*)Kh);
  wmma_gemm<1><<<gg, 256, 0, stream>>>(xh, Wv16, bv, (void*)Vt);

  flash_attn<<<dim3(BATCH * HEADS, SEQ / 128), 256, 0, stream>>>(Qh, Kh, Vt, Ctx);

  wmma_gemm<2><<<gg, 256, 0, stream>>>(Ctx, Wo16, bo, d_out);
}
